// SSBlock_22651657519622
// MI455X (gfx1250) — matrix operand order repaired and checked by reading, not compiled, not run on hardware
//
#include <hip/hip_runtime.h>
#include <math.h>

namespace {

constexpr int kB = 2, kC = 256, kH = 56, kW = 56;
constexpr int kN = kH * kW;       // 3136 tokens per batch
constexpr int kNT = kB * kN;      // 6272 total tokens
constexpr int kHeads = 8, kD = 32;
constexpr int kBS = 8, knW = 7, kNB = 49, kNBT = kB * kNB;  // 98 block rows
constexpr int kHID = 512, kTH = 1024;  // DBFFN hidden / in-proj out
constexpr float kScale = 0.17677669529663689f;  // 32^-0.5

typedef _Float16 half_t;
typedef __attribute__((ext_vector_type(16))) _Float16 v16h;
typedef __attribute__((ext_vector_type(8)))  _Float16 v8h;
typedef __attribute__((ext_vector_type(8)))  float    v8f;

__device__ __forceinline__ float gelu_exact(float x) {
  return 0.5f * x * (1.0f + erff(x * 0.70710678118654752f));
}

__global__ void cvt_f32_f16(const float* __restrict__ s, half_t* __restrict__ d, int n) {
  int i = blockIdx.x * blockDim.x + threadIdx.x;
  if (i < n) d[i] = (half_t)s[i];
}

__device__ __forceinline__ v16h load_a_frag(const half_t* __restrict__ row,
                                            int k0, int hi) {
  v16h a = *(const v16h*)(row + k0 + hi * 16);
  return a;
}

// C[M,N] = A[M,K](f16,row-major) * W[N,K]^T(f16,row-major) + bias. f32 out.
// One wave computes a 32x64 tile: 2 M-subtiles x 4 N-subtiles of 16x16 WMMA
// (v_wmma_f32_16x16x32_f16), A fragments reused 4x, B fragments 2x in-register.
// Requires N % 64 == 0; M tails clamp loads and predicate stores (wave-uniform
// control flow so EXEC is all-ones around WMMA).
// Operand layouts used here (A/B k contiguous per lane half, not the ISA's table order):
//   A 16x32: lanes 0-15 M=lane, elems 0..15 -> K 0..15;
//            lanes 16-31 elems 0..15 -> K 16..31 (same k order as B).
//   B 32x16: lane l (0-15) holds column n0+l, elems = K k0..k0+15; lanes 16-31 K+16.
//   D 16x16 f32: VGPR r -> row hi*8+r, col = lane&15.
__global__ void gemm_f16_wmma(const half_t* __restrict__ A, const half_t* __restrict__ Wt,
                              const float* __restrict__ bias, float* __restrict__ C,
                              int M, int N, int K) {
  int gt = blockIdx.x * blockDim.x + threadIdx.x;
  int wave = gt >> 5;
  int lane = gt & 31;
  int tilesN = N >> 6;           // 64-wide supertiles
  int tilesM = (M + 31) >> 5;    // 32-tall supertiles
  if (wave >= tilesM * tilesN) return;            // wave-uniform exit
  int tm = wave / tilesN;
  int tn = wave - tm * tilesN;
  int l  = lane & 15;
  int hi = lane >> 4;
  int r0 = tm * 32 + l;      if (r0 > M - 1) r0 = M - 1;
  int r1 = tm * 32 + 16 + l; if (r1 > M - 1) r1 = M - 1;
  const half_t* A0 = A + (size_t)r0 * K;
  const half_t* A1 = A + (size_t)r1 * K;
  const half_t* W0 = Wt + (size_t)(tn * 64 + l) * K;
  const half_t* W1 = Wt + (size_t)(tn * 64 + 16 + l) * K;
  const half_t* W2 = Wt + (size_t)(tn * 64 + 32 + l) * K;
  const half_t* W3 = Wt + (size_t)(tn * 64 + 48 + l) * K;
  v8f acc[8];
#pragma unroll
  for (int j = 0; j < 8; ++j) acc[j] = v8f{};
  for (int k0 = 0; k0 < K; k0 += 32) {
    v16h a0 = load_a_frag(A0, k0, hi);
    v16h a1 = load_a_frag(A1, k0, hi);
    v16h b0 = *(const v16h*)(W0 + k0 + hi * 16);
    v16h b1 = *(const v16h*)(W1 + k0 + hi * 16);
    v16h b2 = *(const v16h*)(W2 + k0 + hi * 16);
    v16h b3 = *(const v16h*)(W3 + k0 + hi * 16);
    acc[0] = __builtin_amdgcn_wmma_f32_16x16x32_f16(false, a0, false, b0, (short)0, acc[0], false, false);
    acc[1] = __builtin_amdgcn_wmma_f32_16x16x32_f16(false, a0, false, b1, (short)0, acc[1], false, false);
    acc[2] = __builtin_amdgcn_wmma_f32_16x16x32_f16(false, a0, false, b2, (short)0, acc[2], false, false);
    acc[3] = __builtin_amdgcn_wmma_f32_16x16x32_f16(false, a0, false, b3, (short)0, acc[3], false, false);
    acc[4] = __builtin_amdgcn_wmma_f32_16x16x32_f16(false, a1, false, b0, (short)0, acc[4], false, false);
    acc[5] = __builtin_amdgcn_wmma_f32_16x16x32_f16(false, a1, false, b1, (short)0, acc[5], false, false);
    acc[6] = __builtin_amdgcn_wmma_f32_16x16x32_f16(false, a1, false, b2, (short)0, acc[6], false, false);
    acc[7] = __builtin_amdgcn_wmma_f32_16x16x32_f16(false, a1, false, b3, (short)0, acc[7], false, false);
  }
#pragma unroll
  for (int j = 0; j < 4; ++j) {
    int n = tn * 64 + j * 16 + l;
    float bv = bias ? bias[n] : 0.0f;
#pragma unroll
    for (int r = 0; r < 8; ++r) {
      int m0 = tm * 32 + hi * 8 + r;
      if (m0 < M) C[(size_t)m0 * N + n] = acc[j][r] + bv;
      int m1 = tm * 32 + 16 + hi * 8 + r;
      if (m1 < M) C[(size_t)m1 * N + n] = acc[4 + j][r] + bv;
    }
  }
}

// depthwise 3x3 pos conv (zero pad) + residual + bias; emit NHWC f32 + f16
__global__ void posconv_kernel(const float* __restrict__ x, const float* __restrict__ pw,
                               const float* __restrict__ pb, float* __restrict__ feat,
                               half_t* __restrict__ feath) {
  int i = blockIdx.x * blockDim.x + threadIdx.x;
  if (i >= kNT * kC) return;
  int c = i & (kC - 1);
  int p = i >> 8;
  int xx = p % kW; int p2 = p / kW; int yy = p2 % kH; int b = p2 / kH;
  const float* xb = x + ((size_t)b * kC + c) * kN;   // NCHW plane
  float v = xb[yy * kW + xx] + pb[c];
  for (int dy = 0; dy < 3; ++dy)
    for (int dx = 0; dx < 3; ++dx) {
      int sy = yy + dy - 1, sx = xx + dx - 1;
      if ((unsigned)sy < (unsigned)kH && (unsigned)sx < (unsigned)kW)
        v += xb[sy * kW + sx] * pw[c * 9 + dy * 3 + dx];
    }
  feat[i] = v;
  feath[i] = (half_t)v;
}

__global__ void blockmean_kernel(const float* __restrict__ feat, half_t* __restrict__ blkh) {
  int i = blockIdx.x * blockDim.x + threadIdx.x;
  if (i >= kNBT * kC) return;
  int c = i & (kC - 1);
  int p = i >> 8;
  int nb = p % kNB; int b = p / kNB;
  int by = nb / knW, bx = nb % knW;
  float s = 0.f;
  for (int r = 0; r < kBS; ++r)
    for (int q = 0; q < kBS; ++q)
      s += feat[((size_t)((b * kH + by * kBS + r) * kW) + bx * kBS + q) * kC + c];
  blkh[i] = (half_t)(s * (1.f / 64.f));
}

__global__ void score_kernel(const float* __restrict__ hsc, const float* __restrict__ s2w,
                             const float* __restrict__ s2b, float* __restrict__ scores) {
  int i = blockIdx.x * blockDim.x + threadIdx.x;
  if (i >= kNBT) return;
  float s = s2b[0];
  for (int j = 0; j < 128; ++j) s += gelu_exact(hsc[i * 128 + j]) * s2w[j];
  scores[i] = s;
}

// top-4 per batch, value-desc with lower-index tiebreak (matches lax.top_k)
__global__ void topk_kernel(const float* __restrict__ scores, int* __restrict__ idx) {
  int b = blockIdx.x * blockDim.x + threadIdx.x;
  if (b >= kB) return;
  unsigned long long mask = 0ull;
  for (int k = 0; k < 4; ++k) {
    int best = 0; float bv = -1e30f;
    for (int j = 0; j < kNB; ++j) {
      if (mask & (1ull << j)) continue;
      float v = scores[b * kNB + j];
      if (v > bv) { bv = v; best = j; }
    }
    mask |= (1ull << best);
    idx[b * 4 + k] = best;
  }
}

__global__ void gather_sel_kernel(const half_t* __restrict__ blkh, const int* __restrict__ idx,
                                  half_t* __restrict__ kselh) {
  int i = blockIdx.x * blockDim.x + threadIdx.x;
  if (i >= kB * 4 * kC) return;
  int c = i & (kC - 1);
  int t = i >> 8;
  int b = t >> 2, k = t & 3;
  kselh[i] = blkh[(size_t)(b * kNB + idx[b * 4 + k]) * kC + c];
}

// shared-key attention: K/V rows are [b*nkeys + m]; thread per (b,n,h)
__global__ void attn_kv_kernel(const float* __restrict__ q, const float* __restrict__ k,
                               const float* __restrict__ v, int nkeys,
                               half_t* __restrict__ out) {
  int t = blockIdx.x * blockDim.x + threadIdx.x;
  if (t >= kNT * kHeads) return;
  int h = t & 7;
  int r = t >> 3;              // b*N + n
  int b = r / kN;
  const float* qp = q + (size_t)r * kC + h * kD;
  float s[kNB];
  float mx = -1e30f;
  for (int m = 0; m < nkeys; ++m) {
    const float* kp = k + (size_t)(b * nkeys + m) * kC + h * kD;
    float d = 0.f;
#pragma unroll
    for (int j = 0; j < kD; ++j) d += qp[j] * kp[j];
    d *= kScale;
    s[m] = d;
    mx = fmaxf(mx, d);
  }
  float sum = 0.f;
  for (int m = 0; m < nkeys; ++m) { s[m] = expf(s[m] - mx); sum += s[m]; }
  float inv = 1.f / sum;
  float acc[kD];
#pragma unroll
  for (int j = 0; j < kD; ++j) acc[j] = 0.f;
  for (int m = 0; m < nkeys; ++m) {
    const float* vp = v + (size_t)(b * nkeys + m) * kC + h * kD;
    float w = s[m] * inv;
#pragma unroll
    for (int j = 0; j < kD; ++j) acc[j] += w * vp[j];
  }
  half_t* op = out + (size_t)r * kC + h * kD;
#pragma unroll
  for (int j = 0; j < kD; ++j) op[j] = (half_t)acc[j];
}

// window attention: keys gathered from full projected maps at clamped positions
__global__ void attn_win_kernel(const float* __restrict__ q, const float* __restrict__ kf,
                                const float* __restrict__ vf, half_t* __restrict__ out) {
  int t = blockIdx.x * blockDim.x + threadIdx.x;
  if (t >= kNT * kHeads) return;
  int h = t & 7;
  int r = t >> 3;
  int n = r % kN; int b = r / kN;
  int yy = n / kW, xx = n % kW;
  const float* qp = q + (size_t)r * kC + h * kD;
  float s[49];
  int rows[49];
  float mx = -1e30f;
  for (int m = 0; m < 49; ++m) {
    int dy = m / 7 - 3, dx = m % 7 - 3;
    int sy = yy + dy; sy = sy < 0 ? 0 : (sy > kH - 1 ? kH - 1 : sy);
    int sx = xx + dx; sx = sx < 0 ? 0 : (sx > kW - 1 ? kW - 1 : sx);
    int rr = b * kN + sy * kW + sx;
    rows[m] = rr;
    const float* kp = kf + (size_t)rr * kC + h * kD;
    float d = 0.f;
#pragma unroll
    for (int j = 0; j < kD; ++j) d += qp[j] * kp[j];
    d *= kScale;
    s[m] = d;
    mx = fmaxf(mx, d);
  }
  float sum = 0.f;
  for (int m = 0; m < 49; ++m) { s[m] = expf(s[m] - mx); sum += s[m]; }
  float inv = 1.f / sum;
  float acc[kD];
#pragma unroll
  for (int j = 0; j < kD; ++j) acc[j] = 0.f;
  for (int m = 0; m < 49; ++m) {
    const float* vp = vf + (size_t)rows[m] * kC + h * kD;
    float w = s[m] * inv;
#pragma unroll
    for (int j = 0; j < kD; ++j) acc[j] += w * vp[j];
  }
  half_t* op = out + (size_t)r * kC + h * kD;
#pragma unroll
  for (int j = 0; j < kD; ++j) op[j] = (half_t)acc[j];
}

// LN(Q) -> gate softmax -> fused y = g0*yc + g1*ys + g2*yw + Q (f32 + f16)
__global__ void fuse_kernel(const float* __restrict__ feat, const float* __restrict__ yc,
                            const float* __restrict__ ys, const float* __restrict__ yw,
                            const float* __restrict__ n1w, const float* __restrict__ n1b,
                            const float* __restrict__ gw, const float* __restrict__ gb,
                            float* __restrict__ y, half_t* __restrict__ yh) {
  int r = blockIdx.x * blockDim.x + threadIdx.x;
  if (r >= kNT) return;
  const float* Q = feat + (size_t)r * kC;
  float mu = 0.f;
  for (int c = 0; c < kC; ++c) mu += Q[c];
  mu *= (1.f / kC);
  float var = 0.f;
  for (int c = 0; c < kC; ++c) { float d = Q[c] - mu; var += d * d; }
  var *= (1.f / kC);
  float rstd = rsqrtf(var + 1e-6f);
  float g0 = gb[0], g1 = gb[1], g2 = gb[2];
  for (int c = 0; c < kC; ++c) {
    float qn = (Q[c] - mu) * rstd * n1w[c] + n1b[c];
    g0 += qn * gw[c]; g1 += qn * gw[kC + c]; g2 += qn * gw[2 * kC + c];
  }
  float m = fmaxf(g0, fmaxf(g1, g2));
  float e0 = expf(g0 - m), e1 = expf(g1 - m), e2 = expf(g2 - m);
  float inv = 1.f / (e0 + e1 + e2);
  e0 *= inv; e1 *= inv; e2 *= inv;
  size_t base = (size_t)r * kC;
  for (int c = 0; c < kC; ++c) {
    float v = e0 * yc[base + c] + e1 * ys[base + c] + e2 * yw[base + c] + Q[c];
    y[base + c] = v;
    yh[base + c] = (half_t)v;
  }
}

// depthwise 3x3 on t[.,1024] (zero pad), split, gelu(x1)*x2 -> f16
__global__ void dwgate_kernel(const float* __restrict__ t, const float* __restrict__ dw,
                              half_t* __restrict__ m) {
  int i = blockIdx.x * blockDim.x + threadIdx.x;
  if (i >= kNT * kHID) return;
  int c = i & (kHID - 1);
  int p = i >> 9;
  int xx = p % kW; int p2 = p / kW; int yy = p2 % kH; int b = p2 / kH;
  float u1 = 0.f, u2 = 0.f;
  for (int dy = 0; dy < 3; ++dy)
    for (int dx = 0; dx < 3; ++dx) {
      int sy = yy + dy - 1, sx = xx + dx - 1;
      if ((unsigned)sy < (unsigned)kH && (unsigned)sx < (unsigned)kW) {
        const float* tp = t + (size_t)((b * kH + sy) * kW + sx) * kTH;
        u1 += tp[c] * dw[(dy * 3 + dx) * kTH + c];
        u2 += tp[c + kHID] * dw[(dy * 3 + dx) * kTH + kHID + c];
      }
    }
  m[i] = (half_t)(gelu_exact(u1) * u2);
}

// residual + LN(n2), transpose NHWC -> NCHW output
__global__ void final_kernel(const float* __restrict__ y, const float* __restrict__ mlp,
                             const float* __restrict__ n2w, const float* __restrict__ n2b,
                             float* __restrict__ out) {
  int r = blockIdx.x * blockDim.x + threadIdx.x;
  if (r >= kNT) return;
  int n = r % kN; int b = r / kN;
  int yy = n / kW, xx = n % kW;
  size_t base = (size_t)r * kC;
  float mu = 0.f;
  for (int c = 0; c < kC; ++c) mu += y[base + c] + mlp[base + c];
  mu *= (1.f / kC);
  float var = 0.f;
  for (int c = 0; c < kC; ++c) { float d = y[base + c] + mlp[base + c] - mu; var += d * d; }
  var *= (1.f / kC);
  float rstd = rsqrtf(var + 1e-6f);
  for (int c = 0; c < kC; ++c) {
    float v = (y[base + c] + mlp[base + c] - mu) * rstd * n2w[c] + n2b[c];
    out[((size_t)(b * kC + c) * kH + yy) * kW + xx] = v;
  }
}

}  // namespace

extern "C" void kernel_launch(void* const* d_in, const int* in_sizes, int n_in,
                              void* d_out, int out_size, void* d_ws, size_t ws_size,
                              hipStream_t stream) {
  const float* x      = (const float*)d_in[0];
  const float* pos_w  = (const float*)d_in[1];
  const float* pos_b  = (const float*)d_in[2];
  const float* n1_w   = (const float*)d_in[3];
  const float* n1_b   = (const float*)d_in[4];
  const float* n2_w   = (const float*)d_in[5];
  const float* n2_b   = (const float*)d_in[6];
  const float* wq_w   = (const float*)d_in[7];
  const float* wq_b   = (const float*)d_in[8];
  const float* wk_w   = (const float*)d_in[9];
  const float* wk_b   = (const float*)d_in[10];
  const float* wv_w   = (const float*)d_in[11];
  const float* wv_b   = (const float*)d_in[12];
  const float* pj_w   = (const float*)d_in[13];
  const float* pj_b   = (const float*)d_in[14];
  const float* s1_w   = (const float*)d_in[15];
  const float* s1_b   = (const float*)d_in[16];
  const float* s2_w   = (const float*)d_in[17];
  const float* s2_b   = (const float*)d_in[18];
  const float* g_w    = (const float*)d_in[19];
  const float* g_b    = (const float*)d_in[20];
  const float* pin_w  = (const float*)d_in[21];
  const float* dw_w   = (const float*)d_in[22];
  const float* pout_w = (const float*)d_in[23];
  float* out = (float*)d_out;

  char* wp = (char*)d_ws;
  auto alloc = [&](size_t bytes) -> void* {
    void* r = (void*)wp;
    wp += (bytes + 255) & ~(size_t)255;
    return r;
  };

  const size_t TOK = (size_t)kNT * kC;     // 6272*256
  float*  feat   = (float*)alloc(TOK * 4);
  half_t* feath  = (half_t*)alloc(TOK * 2);
  half_t* blkh   = (half_t*)alloc((size_t)kNBT * kC * 2);
  float*  hscp   = (float*)alloc((size_t)kNBT * 128 * 4);
  float*  scores = (float*)alloc(kNBT * 4);
  int*    selidx = (int*)alloc(kB * 4 * 4);
  half_t* kselh  = (half_t*)alloc((size_t)kB * 4 * kC * 2);
  float*  qcmp   = (float*)alloc(TOK * 4);
  float*  qsel   = (float*)alloc(TOK * 4);
  float*  qwin   = (float*)alloc(TOK * 4);
  float*  kwinF  = (float*)alloc(TOK * 4);
  float*  vwinF  = (float*)alloc(TOK * 4);
  float*  kcmp   = (float*)alloc((size_t)kNBT * kC * 4);
  float*  vcmp   = (float*)alloc((size_t)kNBT * kC * 4);
  float*  kselp  = (float*)alloc((size_t)kB * 4 * kC * 4);
  float*  vselp  = (float*)alloc((size_t)kB * 4 * kC * 4);
  half_t* acmph  = (half_t*)alloc(TOK * 2);
  half_t* aselh  = (half_t*)alloc(TOK * 2);
  half_t* awinh  = (half_t*)alloc(TOK * 2);
  float*  ycmp   = (float*)alloc(TOK * 4);
  float*  ysel   = (float*)alloc(TOK * 4);
  float*  ywin   = (float*)alloc(TOK * 4);
  float*  yf     = (float*)alloc(TOK * 4);
  half_t* yh     = (half_t*)alloc(TOK * 2);
  float*  tbuf   = (float*)alloc((size_t)kNT * kTH * 4);
  half_t* mh     = (half_t*)alloc((size_t)kNT * kHID * 2);
  float*  mlp    = (float*)alloc(TOK * 4);
  half_t* wq_h   = (half_t*)alloc((size_t)3 * kC * kC * 2);
  half_t* wk_h   = (half_t*)alloc((size_t)3 * kC * kC * 2);
  half_t* wv_h   = (half_t*)alloc((size_t)3 * kC * kC * 2);
  half_t* pj_h   = (half_t*)alloc((size_t)3 * kC * kC * 2);
  half_t* s1_h   = (half_t*)alloc((size_t)128 * kC * 2);
  half_t* pin_h  = (half_t*)alloc((size_t)kTH * kC * 2);
  half_t* pout_h = (half_t*)alloc((size_t)kC * kHID * 2);
  (void)ws_size; (void)in_sizes; (void)n_in; (void)out_size;

  auto cvt = [&](const float* s, half_t* d, int n) {
    cvt_f32_f16<<<(n + 255) / 256, 256, 0, stream>>>(s, d, n);
  };
  auto gemm = [&](const half_t* A, const half_t* Wt, const float* bias, float* Cc,
                  int M, int N, int K) {
    long waves = (long)((M + 31) / 32) * (N / 64);
    long threads = waves * 32;
    int block = 128;
    long grid = (threads + block - 1) / block;
    gemm_f16_wmma<<<dim3((unsigned)grid), dim3(block), 0, stream>>>(A, Wt, bias, Cc, M, N, K);
  };

  // weight conversions to f16
  cvt(wq_w, wq_h, 3 * kC * kC);
  cvt(wk_w, wk_h, 3 * kC * kC);
  cvt(wv_w, wv_h, 3 * kC * kC);
  cvt(pj_w, pj_h, 3 * kC * kC);
  cvt(s1_w, s1_h, 128 * kC);
  cvt(pin_w, pin_h, kTH * kC);
  cvt(pout_w, pout_h, kC * kHID);

  // pos conv + residual -> feat (NHWC f32 + f16)
  posconv_kernel<<<(kNT * kC + 255) / 256, 256, 0, stream>>>(x, pos_w, pos_b, feat, feath);

  // block means -> blk f16
  blockmean_kernel<<<(kNBT * kC + 255) / 256, 256, 0, stream>>>(feat, blkh);

  // scorer: hsc_pre = blk @ s1^T + b  (WMMA), then gelu + s2 dot, then top-4
  gemm(blkh, s1_h, s1_b, hscp, kNBT, 128, kC);
  score_kernel<<<1, 128, 0, stream>>>(hscp, s2_w, s2_b, scores);
  topk_kernel<<<1, 32, 0, stream>>>(scores, selidx);
  gather_sel_kernel<<<(kB * 4 * kC + 255) / 256, 256, 0, stream>>>(blkh, selidx, kselh);

  // projections (all WMMA)
  gemm(feath, wq_h,               wq_b,          qcmp,  kNT,    kC, kC);
  gemm(blkh,  wk_h,               wk_b,          kcmp,  kNBT,   kC, kC);
  gemm(blkh,  wv_h,               wv_b,          vcmp,  kNBT,   kC, kC);
  gemm(feath, wq_h + kC * kC,     wq_b + kC,     qsel,  kNT,    kC, kC);
  gemm(kselh, wk_h + kC * kC,     wk_b + kC,     kselp, kB * 4, kC, kC);
  gemm(kselh, wv_h + kC * kC,     wv_b + kC,     vselp, kB * 4, kC, kC);
  gemm(feath, wq_h + 2 * kC * kC, wq_b + 2 * kC, qwin,  kNT,    kC, kC);
  gemm(feath, wk_h + 2 * kC * kC, wk_b + 2 * kC, kwinF, kNT,    kC, kC);  // project-then-gather
  gemm(feath, wv_h + 2 * kC * kC, wv_b + 2 * kC, vwinF, kNT,    kC, kC);

  // attention branches (small key counts -> VALU)
  {
    int total = kNT * kHeads;
    int grid = (total + 127) / 128;
    attn_kv_kernel<<<grid, 128, 0, stream>>>(qcmp, kcmp, vcmp, kNB, acmph);
    attn_kv_kernel<<<grid, 128, 0, stream>>>(qsel, kselp, vselp, 4, aselh);
    attn_win_kernel<<<grid, 128, 0, stream>>>(qwin, kwinF, vwinF, awinh);
  }

  // output projections (WMMA)
  gemm(acmph, pj_h,               pj_b,          ycmp, kNT, kC, kC);
  gemm(aselh, pj_h + kC * kC,     pj_b + kC,     ysel, kNT, kC, kC);
  gemm(awinh, pj_h + 2 * kC * kC, pj_b + 2 * kC, ywin, kNT, kC, kC);

  // gated fusion + residual
  fuse_kernel<<<(kNT + 255) / 256, 256, 0, stream>>>(feat, ycmp, ysel, ywin,
                                                     n1_w, n1_b, g_w, g_b, yf, yh);

  // DBFFN
  gemm(yh, pin_h, nullptr, tbuf, kNT, kTH, kC);
  dwgate_kernel<<<(kNT * kHID + 255) / 256, 256, 0, stream>>>(tbuf, dw_w, mh);
  gemm(mh, pout_h, nullptr, mlp, kNT, kC, kHID);

  // residual + LN + NCHW out
  final_kernel<<<(kNT + 255) / 256, 256, 0, stream>>>(yf, mlp, n2_w, n2_b, out);
}
